// MultiHeadAttention_30374008717799
// MI455X (gfx1250) — matrix operand order repaired and checked by reading, not compiled, not run on hardware
//
#include <hip/hip_runtime.h>

#define DEVI __device__ __forceinline__

typedef __attribute__((ext_vector_type(16))) __bf16 v16bf;
typedef __attribute__((ext_vector_type(2)))  __bf16 v2bf;
typedef __attribute__((ext_vector_type(8)))  float  v8f;

union V16BF { v16bf v; uint4 u[2]; __bf16 h[16]; };

DEVI __bf16 f2bf(float f) { return (__bf16)f; }   // native v_cvt on gfx1250

DEVI float fast_exp2(float x) {
#if __has_builtin(__builtin_amdgcn_exp2f)
  return __builtin_amdgcn_exp2f(x);                // v_exp_f32
#else
  return exp2f(x);
#endif
}

DEVI v2bf pk2(float lo, float hi) {
#if __has_builtin(__builtin_amdgcn_cvt_pk_bf16_f32)
  return __builtin_amdgcn_cvt_pk_bf16_f32(lo, hi);
#else
  v2bf r; r[0] = (__bf16)lo; r[1] = (__bf16)hi; return r;
#endif
}

DEVI v8f wmma_bf16(v16bf a, v16bf b, v8f c) {
  return __builtin_amdgcn_wmma_f32_16x16x32_bf16(
      /*neg_a=*/false, a, /*neg_b=*/false, b,
      /*c_mod=*/(short)0, c, /*reuse_a=*/false, /*reuse_b=*/false);
}

// bf16 fragments (global or LDS pointers; 16B-aligned)
// A (16x32): elems 0..15 = 16 contiguous K at koff
DEVI v16bf loadA_bf(const __bf16* rowp, int koff) {
  V16BF r;
  r.u[0] = *(const uint4*)(rowp + koff);
  r.u[1] = *(const uint4*)(rowp + koff + 8);
  return r.v;
}
// B (32x16): elems 0..15 = 16 contiguous K at koff
DEVI v16bf loadB_bf(const __bf16* rowp, int koff) {
  V16BF r;
  r.u[0] = *(const uint4*)(rowp + koff);
  r.u[1] = *(const uint4*)(rowp + koff + 8);
  return r.v;
}

// f32-source fragments (fallback path only; native packed casts)
DEVI v16bf loadA_f32(const float* rowp, int koff) {
  const float4* p = (const float4*)(rowp + koff);
  float4 a = p[0], b = p[1];
  const float4* q = (const float4*)(rowp + koff + 8);
  float4 c = q[0], d = q[1];
  union { v2bf p[8]; v16bf v; } r;
  r.p[0] = pk2(a.x, a.y); r.p[1] = pk2(a.z, a.w);
  r.p[2] = pk2(b.x, b.y); r.p[3] = pk2(b.z, b.w);
  r.p[4] = pk2(c.x, c.y); r.p[5] = pk2(c.z, c.w);
  r.p[6] = pk2(d.x, d.y); r.p[7] = pk2(d.z, d.w);
  return r.v;
}
DEVI v16bf loadB_f32(const float* rowp, int koff) {
  const float4* p = (const float4*)(rowp + koff);
  float4 a = p[0], b = p[1], c = p[2], d = p[3];
  union { v2bf p[8]; v16bf v; } r;
  r.p[0] = pk2(a.x, a.y); r.p[1] = pk2(a.z, a.w);
  r.p[2] = pk2(b.x, b.y); r.p[3] = pk2(b.z, b.w);
  r.p[4] = pk2(c.x, c.y); r.p[5] = pk2(c.z, c.w);
  r.p[6] = pk2(d.x, d.y); r.p[7] = pk2(d.z, d.w);
  return r.v;
}

// ---------------------------------------------------------------------------
// Kernel 0: bulk f32 -> bf16 conversion (memory-bound pre-pass).
// ---------------------------------------------------------------------------
__global__ __launch_bounds__(256) void cvtbf(const float* __restrict__ src,
                                             __bf16* __restrict__ dst, int n) {
  int i = (blockIdx.x * 256 + threadIdx.x) * 8;
  if (i >= n) return;
  float4 f0 = *(const float4*)(src + i);
  float4 f1 = *(const float4*)(src + i + 4);
  union { v2bf p[4]; uint4 u; } r;
  r.p[0] = pk2(f0.x, f0.y); r.p[1] = pk2(f0.z, f0.w);
  r.p[2] = pk2(f1.x, f1.y); r.p[3] = pk2(f1.z, f1.w);
  *(uint4*)(dst + i) = r.u;
}

// ---------------------------------------------------------------------------
// Kernel 1: q[m, n] = sum_e x[m, e] * Wq[n, e]   (M=8192, N=1024, K=1024)
// bf16 inputs: wave -> 32x64 C tile (2 A frags x 4 B frags = 8 WMMA / k-step).
// ---------------------------------------------------------------------------
__global__ __launch_bounds__(256) void qproj_bf(const __bf16* __restrict__ xb,
                                                const __bf16* __restrict__ wb,
                                                __bf16* __restrict__ qb) {
  const int wave = threadIdx.x >> 5;
  const int lane = threadIdx.x & 31;
  const int lr = lane & 15, hi = lane >> 4;
  const int mb = blockIdx.x >> 4;     // 32 M-blocks of 256
  const int nb = blockIdx.x & 15;     // 16 N-blocks of 64
  const int m0 = mb * 256 + wave * 32;
  const int n0 = nb * 64;

  const __bf16* ar0 = xb + (size_t)(m0 + lr) * 1024;
  const __bf16* ar1 = xb + (size_t)(m0 + 16 + lr) * 1024;
  const __bf16* b0r = wb + (size_t)(n0 +  0 + lr) * 1024;
  const __bf16* b1r = wb + (size_t)(n0 + 16 + lr) * 1024;
  const __bf16* b2r = wb + (size_t)(n0 + 32 + lr) * 1024;
  const __bf16* b3r = wb + (size_t)(n0 + 48 + lr) * 1024;

  v8f acc[2][4];
#pragma unroll
  for (int g = 0; g < 2; ++g)
#pragma unroll
    for (int t = 0; t < 4; ++t) acc[g][t] = (v8f){};

  for (int k0 = 0; k0 < 1024; k0 += 32) {
    const int ka = k0 + hi * 16;
    const int kb = k0 + hi * 16;
    v16bf a0 = loadA_bf(ar0, ka);
    v16bf a1 = loadA_bf(ar1, ka);
    v16bf b0 = loadB_bf(b0r, kb);
    acc[0][0] = wmma_bf16(a0, b0, acc[0][0]);
    acc[1][0] = wmma_bf16(a1, b0, acc[1][0]);
    v16bf b1 = loadB_bf(b1r, kb);
    acc[0][1] = wmma_bf16(a0, b1, acc[0][1]);
    acc[1][1] = wmma_bf16(a1, b1, acc[1][1]);
    v16bf b2 = loadB_bf(b2r, kb);
    acc[0][2] = wmma_bf16(a0, b2, acc[0][2]);
    acc[1][2] = wmma_bf16(a1, b2, acc[1][2]);
    v16bf b3 = loadB_bf(b3r, kb);
    acc[0][3] = wmma_bf16(a0, b3, acc[0][3]);
    acc[1][3] = wmma_bf16(a1, b3, acc[1][3]);
  }
#pragma unroll
  for (int g = 0; g < 2; ++g) {
#pragma unroll
    for (int v = 0; v < 8; ++v) {
      size_t row = (size_t)(m0 + g * 16 + v + 8 * hi) * 1024;
      qb[row + n0 +  0 + lr] = f2bf(acc[g][0][v]);
      qb[row + n0 + 16 + lr] = f2bf(acc[g][1][v]);
      qb[row + n0 + 32 + lr] = f2bf(acc[g][2][v]);
      qb[row + n0 + 48 + lr] = f2bf(acc[g][3][v]);
    }
  }
}

// Fallback: on-the-fly conversion (used only if ws too small for bf16 copies)
__global__ __launch_bounds__(256) void qproj_f32(const float* __restrict__ x,
                                                 const float* __restrict__ Wq,
                                                 __bf16* __restrict__ qb) {
  const int wave = threadIdx.x >> 5;
  const int lane = threadIdx.x & 31;
  const int lr = lane & 15, hi = lane >> 4;
  const int mb = blockIdx.x >> 4;
  const int nb = blockIdx.x & 15;
  const int m0 = mb * 128 + wave * 16;
  const int n0 = nb * 64;
  const float* arow = x + (size_t)(m0 + lr) * 1024;
  const float* b0r = Wq + (size_t)(n0 +  0 + lr) * 1024;
  const float* b1r = Wq + (size_t)(n0 + 16 + lr) * 1024;
  const float* b2r = Wq + (size_t)(n0 + 32 + lr) * 1024;
  const float* b3r = Wq + (size_t)(n0 + 48 + lr) * 1024;
  v8f acc0 = {}, acc1 = {}, acc2 = {}, acc3 = {};
  for (int k0 = 0; k0 < 1024; k0 += 32) {
    v16bf a = loadA_f32(arow, k0 + hi * 16);
    const int kb = k0 + hi * 16;
    acc0 = wmma_bf16(a, loadB_f32(b0r, kb), acc0);
    acc1 = wmma_bf16(a, loadB_f32(b1r, kb), acc1);
    acc2 = wmma_bf16(a, loadB_f32(b2r, kb), acc2);
    acc3 = wmma_bf16(a, loadB_f32(b3r, kb), acc3);
  }
#pragma unroll
  for (int v = 0; v < 8; ++v) {
    size_t row = (size_t)(m0 + v + 8 * hi) * 1024;
    qb[row + n0 +  0 + lr] = f2bf(acc0[v]);
    qb[row + n0 + 16 + lr] = f2bf(acc1[v]);
    qb[row + n0 + 32 + lr] = f2bf(acc2[v]);
    qb[row + n0 + 48 + lr] = f2bf(acc3[v]);
  }
}

// ---------------------------------------------------------------------------
// Kernel 2: flash-style causal attention, q == k == v (bf16), accum f32.
// One workgroup (8 waves) per (b, h, 128-query block). Wave w owns 16 rows.
// Online softmax entirely in the C-fragment register layout.
// ---------------------------------------------------------------------------
#define SLEN 2048
#define VSTR 136   // padded bf16 row stride (272B = 17 x 16B -> conflict-free)

__global__ __launch_bounds__(256) void attn(const __bf16* __restrict__ qb,
                                            float* __restrict__ out) {
  __shared__ __bf16 sVt[64 * VSTR];        // transposed K/V block: Vt[d][k]
  __shared__ __bf16 sP[8 * 16 * VSTR];     // per-wave P rows (C->A relayout)

  const int wave = threadIdx.x >> 5;
  const int lane = threadIdx.x & 31;
  const int lr = lane & 15, hi = lane >> 4;
  const int qi = blockIdx.x & 15;          // query block index (128 rows)
  const int h  = (blockIdx.x >> 4) & 15;
  const int b  = blockIdx.x >> 8;
  const int qb0 = qi * 128;
  const __bf16* qhead = qb + (size_t)b * SLEN * 1024 + h * 64;

  const __bf16* qrowp = qhead + (size_t)(qb0 + wave * 16 + lr) * 1024;
  const v16bf aQ0 = loadA_bf(qrowp, 0  + hi * 16);
  const v16bf aQ1 = loadA_bf(qrowp, 32 + hi * 16);

  v8f o0 = {}, o1 = {}, o2 = {}, o3 = {};
  float mrow[8], lrow[8];
#pragma unroll
  for (int v = 0; v < 8; ++v) { mrow[v] = -__builtin_inff(); lrow[v] = 0.0f; }
  // softmax(s / 32): fold scale into exp2 -> p = 2^((s - m) * log2(e)/32)
  const float c2 = 0.04508422f;            // log2(e) / 32

  __bf16* pw = sP + wave * 16 * VSTR;

  for (int kbI = 0; kbI <= qi; ++kbI) {
    const int kb0 = kbI * 128;
    __syncthreads();                       // previous PV done reading sVt
    {  // stage transposed V block: thread -> key k, 32 dims
      const int k  = threadIdx.x & 127;
      const int d0 = (threadIdx.x >> 7) * 32;
      const __bf16* src = qhead + (size_t)(kb0 + k) * 1024 + d0;
      union { uint4 u[4]; __bf16 h[32]; } cv;
      cv.u[0] = *(const uint4*)(src);
      cv.u[1] = *(const uint4*)(src + 8);
      cv.u[2] = *(const uint4*)(src + 16);
      cv.u[3] = *(const uint4*)(src + 24);
#pragma unroll
      for (int j = 0; j < 32; ++j) sVt[(d0 + j) * VSTR + k] = cv.h[j];
    }
    __syncthreads();

    // --- scores: S = Q * K^T  (8 tiles of 16x16, K-dim = 64 = 2 WMMAs) ---
    v8f s[8];
#pragma unroll
    for (int t = 0; t < 8; ++t) {
      const __bf16* krow = qhead + (size_t)(kb0 + t * 16 + lr) * 1024;
      v8f z = {};
      z    = wmma_bf16(aQ0, loadB_bf(krow, 0  + hi * 16), z);
      s[t] = wmma_bf16(aQ1, loadB_bf(krow, 32 + hi * 16), z);
    }

    // --- causal mask (diagonal block only) + running row max ---
    // C layout: row = v + 8*hi (per VGPR), col/key = lr (per lane).
    float mnew[8];
#pragma unroll
    for (int v = 0; v < 8; ++v) mnew[v] = mrow[v];

    if (kbI == qi) {                       // uniform branch: diagonal block
      const int qrow0 = qb0 + wave * 16 + 8 * hi;
#pragma unroll
      for (int t = 0; t < 8; ++t) {
        const int kcol = kb0 + t * 16 + lr;
#pragma unroll
        for (int v = 0; v < 8; ++v) {
          float sv = (kcol <= qrow0 + v) ? s[t][v] : -__builtin_inff();
          s[t][v] = sv;
          mnew[v] = fmaxf(mnew[v], sv);
        }
      }
    } else {                               // fully-unmasked block: no selects
#pragma unroll
      for (int t = 0; t < 8; ++t)
#pragma unroll
        for (int v = 0; v < 8; ++v) mnew[v] = fmaxf(mnew[v], s[t][v]);
    }

#pragma unroll
    for (int v = 0; v < 8; ++v) {          // 16-lane (half-wave) max tree
      float m = mnew[v];
      m = fmaxf(m, __shfl_xor(m, 1, 32));
      m = fmaxf(m, __shfl_xor(m, 2, 32));
      m = fmaxf(m, __shfl_xor(m, 4, 32));
      m = fmaxf(m, __shfl_xor(m, 8, 32));
      mnew[v] = m;
    }

    float alpha[8], rsum[8], negem[8];
#pragma unroll
    for (int v = 0; v < 8; ++v) {
      alpha[v] = fast_exp2((mrow[v] - mnew[v]) * c2);
      negem[v] = -mnew[v] * c2;
      mrow[v] = mnew[v];
      rsum[v] = 0.0f;
    }

    // --- P = 2^(s*c2 - m*c2); spill P (bf16) to LDS in A-fragment layout ---
#pragma unroll
    for (int t = 0; t < 8; ++t) {
#pragma unroll
      for (int v = 0; v < 8; ++v) {
        float p = fast_exp2(__builtin_fmaf(s[t][v], c2, negem[v]));
        rsum[v] += p;
        pw[(v + 8 * hi) * VSTR + t * 16 + lr] = f2bf(p);
      }
    }
#pragma unroll
    for (int v = 0; v < 8; ++v) {          // 16-lane sum tree + rescale O
      float r = rsum[v];
      r += __shfl_xor(r, 1, 32);
      r += __shfl_xor(r, 2, 32);
      r += __shfl_xor(r, 4, 32);
      r += __shfl_xor(r, 8, 32);
      lrow[v] = lrow[v] * alpha[v] + r;
      o0[v] *= alpha[v]; o1[v] *= alpha[v];
      o2[v] *= alpha[v]; o3[v] *= alpha[v];
    }

    // --- O += P * V  (K-dim = 128 keys, 4 WMMA steps x 4 d-tiles) ---
#pragma unroll
    for (int kk = 0; kk < 128; kk += 32) {
      v16bf aP = loadA_bf(pw + lr * VSTR, kk + hi * 16);
      o0 = wmma_bf16(aP, loadB_bf(sVt + ( 0 + lr) * VSTR, kk + hi * 16), o0);
      o1 = wmma_bf16(aP, loadB_bf(sVt + (16 + lr) * VSTR, kk + hi * 16), o1);
      o2 = wmma_bf16(aP, loadB_bf(sVt + (32 + lr) * VSTR, kk + hi * 16), o2);
      o3 = wmma_bf16(aP, loadB_bf(sVt + (48 + lr) * VSTR, kk + hi * 16), o3);
    }
  }

  // --- epilogue: normalize and store ctx[b, s, h*64 + d] (f32) ---
#pragma unroll
  for (int v = 0; v < 8; ++v) {
    const float rinv = 1.0f / lrow[v];
    size_t row = ((size_t)b * SLEN + qb0 + wave * 16 + v + 8 * hi) * 1024
               + (size_t)h * 64;
    out[row +  0 + lr] = o0[v] * rinv;
    out[row + 16 + lr] = o1[v] * rinv;
    out[row + 32 + lr] = o2[v] * rinv;
    out[row + 48 + lr] = o3[v] * rinv;
  }
}

extern "C" void kernel_launch(void* const* d_in, const int* in_sizes, int n_in,
                              void* d_out, int out_size, void* d_ws, size_t ws_size,
                              hipStream_t stream) {
  const float* x  = (const float*)d_in[0];   // [4, 2048, 1024] f32
  const float* Wq = (const float*)d_in[1];   // [1024, 1024] f32
  float* out = (float*)d_out;                // [4, 2048, 1024] f32

  const size_t QBYTES = (size_t)8192 * 1024 * 2;   // q bf16, 16 MB
  const size_t XBYTES = (size_t)8192 * 1024 * 2;   // x bf16, 16 MB
  const size_t WBYTES = (size_t)1024 * 1024 * 2;   // Wq bf16, 2 MB

  __bf16* qbuf = (__bf16*)d_ws;

  if (ws_size >= QBYTES + XBYTES + WBYTES) {
    __bf16* xbf = (__bf16*)((char*)d_ws + QBYTES);
    __bf16* wbf = (__bf16*)((char*)d_ws + QBYTES + XBYTES);
    cvtbf<<<dim3(8388608 / 2048), dim3(256), 0, stream>>>(x,  xbf, 8388608);
    cvtbf<<<dim3(1048576 / 2048), dim3(256), 0, stream>>>(Wq, wbf, 1048576);
    qproj_bf<<<dim3(32 * 16), dim3(256), 0, stream>>>(xbf, wbf, qbuf);
  } else {
    qproj_f32<<<dim3(64 * 16), dim3(256), 0, stream>>>(x, Wq, qbuf);
  }
  attn<<<dim3(4 * 16 * 16), dim3(256), 0, stream>>>(qbuf, out);
}
